// Encoder_84954453114903
// MI455X (gfx1250) — matrix operand order repaired and checked by reading, not compiled, not run on hardware
//
#include <hip/hip_runtime.h>
#include <hip/hip_bf16.h>

// ---------------- problem constants ----------------
#define BB   8
#define SS   1024
#define DD   512
#define HH   8
#define LL   6
#define DFF  1024
#define DH   64
#define MM   (BB*SS)          // 8192 rows

typedef __attribute__((ext_vector_type(16))) _Float16 v16h;
typedef __attribute__((ext_vector_type(8)))  _Float16 v8h;
typedef __attribute__((ext_vector_type(8)))  float    v8f;

static __device__ __forceinline__ v8f wmma_f16(v16h a, v16h b, v8f c) {
  return __builtin_amdgcn_wmma_f32_16x16x32_f16(false, a, false, b, (short)0, c, false, false);
}

// CDNA5 async global->LDS copy (16B per lane), tracked by ASYNCcnt.
static __device__ __forceinline__ void async_copy_b128(void* lds_dst, const void* gsrc) {
  unsigned int  laddr = (unsigned int)(unsigned long long)lds_dst; // LDS offset = low 32 bits
  unsigned long long ga = (unsigned long long)gsrc;
  asm volatile("global_load_async_to_lds_b128 %0, %1, off" :: "v"(laddr), "v"(ga) : "memory");
}
static __device__ __forceinline__ void wait_asynccnt0() {
  asm volatile("s_wait_asynccnt 0" ::: "memory");
}

// ---------------- weight packing (f32 KxN -> f16 WMMA-B fragment layout) ----
// dst[((tn*ntk + tk)*32 + lane)*16 + h] = W[(tk*32 + 16*(lane>>4) + h)*N + tn*16 + (lane&15)]
__global__ void pack_b_kernel(const float* __restrict__ W, _Float16* __restrict__ out,
                              int K, int N) {
  int tid = blockIdx.x * 256 + threadIdx.x;
  if (tid >= K * N) return;
  int ntk = K >> 5;
  int h    = tid & 15;
  int lane = (tid >> 4) & 31;
  int tile = tid >> 9;
  int tk = tile % ntk;
  int tn = tile / ntk;
  int k = tk * 32 + 16 * (lane >> 4) + h;
  int n = tn * 16 + (lane & 15);
  out[tid] = (_Float16)W[(size_t)k * N + n];
}

// ---------------- embedding + positional encoding (writes f32 + f16 mirror) ----
__global__ void embed_kernel(const int* __restrict__ x, const float* __restrict__ emb,
                             float* __restrict__ h, _Float16* __restrict__ h16) {
  int row = blockIdx.x;                 // 0..MM-1
  int s   = row & (SS - 1);
  int tok = x[row];
  const float kln = 9.210340371976184f / (float)DD;   // ln(10000)/D
  for (int d = threadIdx.x; d < DD; d += 256) {
    float e = emb[(size_t)tok * DD + d] * 22.62741699796952f;  // sqrt(512)
    int i = d >> 1;
    float ang = (float)s * __expf(-(float)(2 * i) * kln);
    float pe = (d & 1) ? __cosf(ang) : __sinf(ang);
    float v = e + pe;
    h[(size_t)row * DD + d]   = v;
    h16[(size_t)row * DD + d] = (_Float16)v;
  }
}

// ---------------- WMMA GEMM: C[MxN] = A[MxK](f16 row-major) @ Bp(f16 packed) + bias
// 256 threads = 8 waves; each wave computes a 64x32 strip (4 m-tiles x 2 n-tiles).
// The block's whole B strip (2 n-tiles x K) is async-copied to LDS once and
// shared by all 8 waves (8x cut in global B traffic).
template<typename OutT, bool RELU>
__global__ void gemm_kernel(const _Float16* __restrict__ A, const _Float16* __restrict__ Bp,
                            const float* __restrict__ bias, OutT* __restrict__ C,
                            int M, int N, int K) {
  extern __shared__ _Float16 bsh[];               // 2 * ntk * 512 halves
  const int lane = threadIdx.x & 31;
  const int w    = threadIdx.x >> 5;
  const int hf   = lane >> 4;
  const int ln16 = lane & 15;
  const int mbase = blockIdx.x * 512 + w * 64;
  const int tn0 = blockIdx.y * 2;          // two consecutive 16-col tiles
  const int ntk = K >> 5;

  // ---- async stage of packed B strip into LDS ----
  {
    const char* gsrc = (const char*)(Bp + (size_t)tn0 * ntk * 512);
    char* lbase = (char*)bsh;
    const int nchunk = ntk * 128;                 // 16-byte chunks (strip = ntk*2048 B)
    for (int c = threadIdx.x; c < nchunk; c += 256)
      async_copy_b128(lbase + c * 16, gsrc + c * 16);
    wait_asynccnt0();
    __syncthreads();
  }

  v8f zero = {0.f,0.f,0.f,0.f,0.f,0.f,0.f,0.f};
  v8f acc[4][2] = {{zero,zero},{zero,zero},{zero,zero},{zero,zero}};

  const _Float16* bl0 = bsh + lane * 16;                      // tile tn0
  const _Float16* bl1 = bsh + (size_t)ntk * 512 + lane * 16;  // tile tn0+1

  for (int tk = 0; tk < ntk; ++tk) {
    v16h bf0 = *(const v16h*)(bl0 + (size_t)tk * 512);
    v16h bf1 = *(const v16h*)(bl1 + (size_t)tk * 512);
    const int k0 = tk * 32;
#pragma unroll
    for (int t = 0; t < 4; ++t) {
      const _Float16* ar = A + (size_t)(mbase + t * 16 + ln16) * K + k0 + 16 * hf;
      v8h lo = *(const v8h*)ar;
      v8h hi = *(const v8h*)(ar + 8);
      v16h af;
#pragma unroll
      for (int i = 0; i < 8; ++i) { af[i] = lo[i]; af[8 + i] = hi[i]; }
      acc[t][0] = wmma_f16(af, bf0, acc[t][0]);
      acc[t][1] = wmma_f16(af, bf1, acc[t][1]);
    }
  }

  float bv0 = bias[tn0 * 16 + ln16];
  float bv1 = bias[tn0 * 16 + 16 + ln16];
#pragma unroll
  for (int t = 0; t < 4; ++t) {
#pragma unroll
    for (int r = 0; r < 8; ++r) {
      size_t rowoff = (size_t)(mbase + t * 16 + r + 8 * hf) * N;
      float v0 = acc[t][0][r] + bv0;
      float v1 = acc[t][1][r] + bv1;
      if (RELU) { v0 = fmaxf(v0, 0.0f); v1 = fmaxf(v1, 0.0f); }
      C[rowoff + tn0 * 16 + ln16]      = (OutT)v0;
      C[rowoff + tn0 * 16 + 16 + ln16] = (OutT)v1;
    }
  }
}

// ---------------- V transpose: f16 [B,S,D] -> f16 [B*H, DH, S] ----------------
__global__ void vtrans_kernel(const _Float16* __restrict__ V, _Float16* __restrict__ Vt) {
  size_t tid = (size_t)blockIdx.x * 256 + threadIdx.x;
  if (tid >= (size_t)BB * SS * DD) return;
  int s = tid & (SS - 1);
  size_t rest = tid >> 10;
  int dl = rest & (DH - 1);
  int bh = (int)(rest >> 6);
  int b = bh >> 3, hh = bh & 7;
  Vt[tid] = V[((size_t)(b * SS + s)) * DD + hh * DH + dl];
}

// ---------------- fused flash attention (WMMA logits + WMMA P@V) ----------------
__global__ void attn_kernel(const _Float16* __restrict__ Qh, const _Float16* __restrict__ Kh,
                            const _Float16* __restrict__ Vt, const unsigned char* __restrict__ mask,
                            _Float16* __restrict__ Oout) {
  __shared__ _Float16 plds[4 * 16 * 32];
  const int lane = threadIdx.x & 31;
  const int w    = threadIdx.x >> 5;
  const int hf   = lane >> 4;
  const int ln16 = lane & 15;
  const int bh = blockIdx.y;
  const int b  = bh >> 3, hh = bh & 7;
  const int q0 = blockIdx.x * 64 + w * 16;

  // Q fragments (scale 1/sqrt(64)=0.125 folded in)
  v16h aQ[2];
  const _Float16* qrow = Qh + ((size_t)(b * SS + q0 + ln16)) * DD + hh * DH;
#pragma unroll
  for (int t = 0; t < 2; ++t) {
    const _Float16* p = qrow + t * 32 + 16 * hf;
    v8h lo = *(const v8h*)p;
    v8h hi = *(const v8h*)(p + 8);
#pragma unroll
    for (int i = 0; i < 8; ++i) {
      aQ[t][i]     = lo[i] * (_Float16)0.125f;
      aQ[t][8 + i] = hi[i] * (_Float16)0.125f;
    }
  }

  float mrow[8], lrow[8];
  v8f zero = {0.f,0.f,0.f,0.f,0.f,0.f,0.f,0.f};
  v8f o[4] = {zero, zero, zero, zero};
#pragma unroll
  for (int r = 0; r < 8; ++r) { mrow[r] = -1.0e30f; lrow[r] = 0.0f; }

  _Float16* myp = plds + w * 512;

  for (int kb = 0; kb < SS; kb += 32) {
    v8f c[2];
#pragma unroll
    for (int kt = 0; kt < 2; ++kt) {
      const int krow = kb + kt * 16 + ln16;
      const _Float16* kp = Kh + ((size_t)(b * SS + krow)) * DD + hh * DH + 16 * hf;
      v16h b0 = *(const v16h*)kp;
      v16h b1 = *(const v16h*)(kp + 32);
      v8f cc = zero;
      cc = wmma_f16(aQ[0], b0, cc);
      cc = wmma_f16(aQ[1], b1, cc);
      c[kt] = cc;
    }
    // key mask (broadcast over q rows)
#pragma unroll
    for (int kt = 0; kt < 2; ++kt) {
      if (mask[b * SS + kb + kt * 16 + ln16]) {
#pragma unroll
        for (int r = 0; r < 8; ++r) c[kt][r] = -1.0e9f;
      }
    }
    // row max over 16-lane halves
    float rm[8];
#pragma unroll
    for (int r = 0; r < 8; ++r) rm[r] = fmaxf(c[0][r], c[1][r]);
    for (int d = 1; d < 16; d <<= 1)
#pragma unroll
      for (int r = 0; r < 8; ++r) rm[r] = fmaxf(rm[r], __shfl_xor(rm[r], d));

    float esc[8], rs[8];
#pragma unroll
    for (int r = 0; r < 8; ++r) {
      float mn = fmaxf(mrow[r], rm[r]);
      esc[r] = __expf(mrow[r] - mn);
      mrow[r] = mn;
      c[0][r] = __expf(c[0][r] - mn);
      c[1][r] = __expf(c[1][r] - mn);
      rs[r] = c[0][r] + c[1][r];
    }
    for (int d = 1; d < 16; d <<= 1)
#pragma unroll
      for (int r = 0; r < 8; ++r) rs[r] += __shfl_xor(rs[r], d);
#pragma unroll
    for (int r = 0; r < 8; ++r) lrow[r] = lrow[r] * esc[r] + rs[r];
#pragma unroll
    for (int t = 0; t < 4; ++t)
#pragma unroll
      for (int r = 0; r < 8; ++r) o[t][r] *= esc[r];

    // C-layout P -> LDS -> A-layout f16 fragment (16x32)
    __syncthreads();
#pragma unroll
    for (int kt = 0; kt < 2; ++kt)
#pragma unroll
      for (int r = 0; r < 8; ++r)
        myp[(r + 8 * hf) * 32 + kt * 16 + ln16] = (_Float16)c[kt][r];
    __syncthreads();

    v16h aP;
    {
      const _Float16* pr = myp + ln16 * 32 + 16 * hf;
      v8h lo = *(const v8h*)pr;
      v8h hi = *(const v8h*)(pr + 8);
#pragma unroll
      for (int i = 0; i < 8; ++i) { aP[i] = lo[i]; aP[8 + i] = hi[i]; }
    }

    // O += P @ V  (V pre-transposed: contiguous keys at fixed d)
#pragma unroll
    for (int t = 0; t < 4; ++t) {
      const _Float16* vp = Vt + ((size_t)(bh * DH + t * 16 + ln16)) * SS + kb + 16 * hf;
      v16h bv = *(const v16h*)vp;
      o[t] = wmma_f16(aP, bv, o[t]);
    }
  }

  // epilogue: normalize and store f16 back to [B,S,D] with head-major columns
#pragma unroll
  for (int r = 0; r < 8; ++r) {
    float inv = 1.0f / lrow[r];
#pragma unroll
    for (int t = 0; t < 4; ++t) {
      Oout[((size_t)(b * SS + q0 + r + 8 * hf)) * DD + hh * DH + t * 16 + ln16] =
          (_Float16)(o[t][r] * inv);
    }
  }
}

// ---------------- residual + LayerNorm (writes f32 + f16 mirror) ----------------
__device__ __forceinline__ float block_sum(float v, float* sm) {
  for (int d = 16; d > 0; d >>= 1) v += __shfl_xor(v, d);
  if ((threadIdx.x & 31) == 0) sm[threadIdx.x >> 5] = v;
  __syncthreads();
  if (threadIdx.x == 0) {
    float t = 0.f;
    for (int i = 0; i < 8; ++i) t += sm[i];
    sm[0] = t;
  }
  __syncthreads();
  float r = sm[0];
  __syncthreads();
  return r;
}

__global__ void add_ln_kernel(const float* __restrict__ x, const float* __restrict__ res,
                              const float* __restrict__ g, const float* __restrict__ be,
                              float* __restrict__ out, _Float16* __restrict__ out16) {
  __shared__ float sm[8];
  int row = blockIdx.x;
  int t0 = threadIdx.x, t1 = threadIdx.x + 256;
  float a0 = x[(size_t)row * DD + t0] + res[(size_t)row * DD + t0];
  float a1 = x[(size_t)row * DD + t1] + res[(size_t)row * DD + t1];
  float mu = block_sum(a0 + a1, sm) * (1.0f / DD);
  float d0 = a0 - mu, d1 = a1 - mu;
  float var = block_sum(d0 * d0 + d1 * d1, sm) * (1.0f / DD);
  float rstd = rsqrtf(var + 1e-5f);
  float o0 = d0 * rstd * g[t0] + be[t0];
  float o1 = d1 * rstd * g[t1] + be[t1];
  out[(size_t)row * DD + t0] = o0;
  out[(size_t)row * DD + t1] = o1;
  out16[(size_t)row * DD + t0] = (_Float16)o0;
  out16[(size_t)row * DD + t1] = (_Float16)o1;
}

// ---------------- host side ----------------
extern "C" void kernel_launch(void* const* d_in, const int* in_sizes, int n_in,
                              void* d_out, int out_size, void* d_ws, size_t ws_size,
                              hipStream_t stream) {
  const int*           x    = (const int*)d_in[0];
  const unsigned char* mask = (const unsigned char*)d_in[1];
  const float* emb = (const float*)d_in[2];
  const float* Wq  = (const float*)d_in[3];
  const float* bq  = (const float*)d_in[4];
  const float* Wk  = (const float*)d_in[5];
  const float* bk  = (const float*)d_in[6];
  const float* Wv  = (const float*)d_in[7];
  const float* bv  = (const float*)d_in[8];
  const float* Wo  = (const float*)d_in[9];
  const float* bo  = (const float*)d_in[10];
  const float* W1  = (const float*)d_in[11];
  const float* b1  = (const float*)d_in[12];
  const float* W2  = (const float*)d_in[13];
  const float* b2  = (const float*)d_in[14];
  const float* g1  = (const float*)d_in[15];
  const float* be1 = (const float*)d_in[16];
  const float* g2  = (const float*)d_in[17];
  const float* be2 = (const float*)d_in[18];

  // packed-weight per-layer layout (in halves)
  const size_t PL   = 4 * (size_t)DD * DD + 2 * (size_t)DD * DFF;  // 2,097,152
  const size_t OQ = 0, OK = (size_t)DD * DD, OV = 2 * OK, OO = 3 * OK;
  const size_t O1 = 4 * OK, O2 = O1 + (size_t)DD * DFF;

  char* ws = (char*)d_ws;
  _Float16* pw  = (_Float16*)ws;                         // 24 MB packed weights
  size_t off = LL * PL * sizeof(_Float16);
  float*    h     = (float*)(ws + off);    off += (size_t)MM * DD * 4;   // 16 MB
  _Float16* h16   = (_Float16*)(ws + off); off += (size_t)MM * DD * 2;   //  8 MB
  _Float16* qh    = (_Float16*)(ws + off); off += (size_t)MM * DD * 2;   //  8 MB
  _Float16* kh    = (_Float16*)(ws + off); off += (size_t)MM * DD * 2;   //  8 MB
  _Float16* vh    = (_Float16*)(ws + off); off += (size_t)MM * DD * 2;   //  8 MB
  _Float16* vt    = (_Float16*)(ws + off); off += (size_t)MM * DD * 2;   //  8 MB
  _Float16* att16 = (_Float16*)(ws + off); off += (size_t)MM * DD * 2;   //  8 MB
  _Float16* ff16  = (_Float16*)(ws + off); off += (size_t)MM * DFF * 2;  // 16 MB
  float*    tmpB  = (float*)(ws + off);    off += (size_t)MM * DD * 4;   // 16 MB

  // ---- pack all weights (idempotent, deterministic) ----
  for (int l = 0; l < LL; ++l) {
    int nDD  = DD * DD, nDF = DD * DFF;
    pack_b_kernel<<<(nDD + 255) / 256, 256, 0, stream>>>(Wq + (size_t)l * nDD, pw + l * PL + OQ, DD, DD);
    pack_b_kernel<<<(nDD + 255) / 256, 256, 0, stream>>>(Wk + (size_t)l * nDD, pw + l * PL + OK, DD, DD);
    pack_b_kernel<<<(nDD + 255) / 256, 256, 0, stream>>>(Wv + (size_t)l * nDD, pw + l * PL + OV, DD, DD);
    pack_b_kernel<<<(nDD + 255) / 256, 256, 0, stream>>>(Wo + (size_t)l * nDD, pw + l * PL + OO, DD, DD);
    pack_b_kernel<<<(nDF + 255) / 256, 256, 0, stream>>>(W1 + (size_t)l * nDF, pw + l * PL + O1, DD, DFF);
    pack_b_kernel<<<(nDF + 255) / 256, 256, 0, stream>>>(W2 + (size_t)l * nDF, pw + l * PL + O2, DFF, DD);
  }

  // ---- embedding + positional encoding ----
  embed_kernel<<<MM, 256, 0, stream>>>(x, emb, h, h16);

  dim3 gD(MM / 512, DD / 32);    // GEMM -> N=512  (wave strip 64x32)
  dim3 gF(MM / 512, DFF / 32);   // GEMM -> N=1024
  dim3 gAttn(SS / 64, BB * HH);
  const size_t ldsD = (DD / 32) * 2048;    // B strip LDS bytes, K=512  -> 32 KB
  const size_t ldsF = (DFF / 32) * 2048;   // K=1024 -> 64 KB

  for (int l = 0; l < LL; ++l) {
    const _Float16* pl = pw + l * PL;
    // QKV projections (all f16 out)
    gemm_kernel<_Float16, false><<<gD, 256, ldsD, stream>>>(h16, pl + OQ, bq + l * DD, qh, MM, DD, DD);
    gemm_kernel<_Float16, false><<<gD, 256, ldsD, stream>>>(h16, pl + OK, bk + l * DD, kh, MM, DD, DD);
    gemm_kernel<_Float16, false><<<gD, 256, ldsD, stream>>>(h16, pl + OV, bv + l * DD, vh, MM, DD, DD);
    vtrans_kernel<<<((size_t)MM * DD + 255) / 256, 256, 0, stream>>>(vh, vt);
    // fused attention -> f16 [B,S,D]
    attn_kernel<<<gAttn, 128, 0, stream>>>(qh, kh, vt, mask, att16);
    // output projection + residual LN
    gemm_kernel<float, false><<<gD, 256, ldsD, stream>>>(att16, pl + OO, bo + l * DD, tmpB, MM, DD, DD);
    add_ln_kernel<<<MM, 256, 0, stream>>>(h, tmpB, g1 + l * DD, be1 + l * DD, h, h16);
    // FFN
    gemm_kernel<_Float16, true ><<<gF, 256, ldsD, stream>>>(h16, pl + O1, b1 + l * DFF, ff16, MM, DFF, DD);
    gemm_kernel<float,    false><<<gD, 256, ldsF, stream>>>(ff16, pl + O2, b2 + l * DD, tmpB, MM, DD, DFF);
    add_ln_kernel<<<MM, 256, 0, stream>>>(h, tmpB, g2 + l * DD, be2 + l * DD, h, h16);
  }

  hipMemcpyAsync(d_out, h, (size_t)MM * DD * sizeof(float), hipMemcpyDeviceToDevice, stream);
}
